// ViewDependentColorMLP_25555055411357
// MI455X (gfx1250) — matrix operand order repaired and checked by reading, not compiled, not run on hardware
//
#include <hip/hip_runtime.h>
#include <hip/hip_bf16.h>

typedef __attribute__((ext_vector_type(16))) _Float16 v16h;
typedef __attribute__((ext_vector_type(8)))  float    v8f;

#define NSPL   300000
#define NCAMS  8
#define FDIM   32
#define EDIM   16
#define WID    64
#define TILES_PER_CAM (NSPL / 16)
#define NTILES (NCAMS * TILES_PER_CAM)
#define HSTR   72                 // halves per activation row in LDS (16B-aligned stride)
#define NFRAG  18                 // 8 (w0) + 8 (w1) + 2 (w2 padded to N=16)
#define FRAG_HALVES 512           // 32 lanes * 16 halves

// ---------------------------------------------------------------------------
// Pack fp32 weights into WMMA f16 B-fragments (lane-major, per CDNA5 ISA):
//   lane L holds column (L&15); K rows: lanes 16-31 take K+16; half j -> K=j.
//   Fragment (k,n) covers B[k*32 .. +31][n*16 .. +15].
// ---------------------------------------------------------------------------
__global__ void prep_weights_kernel(const float* __restrict__ w0,
                                    const float* __restrict__ w1,
                                    const float* __restrict__ w2,
                                    _Float16* __restrict__ frags) {
  int idx = blockIdx.x * blockDim.x + threadIdx.x;
  if (idx >= NFRAG * FRAG_HALVES) return;
  int f = idx >> 9;            // fragment id
  int e = idx & 511;           // element within fragment
  int L = e >> 4;              // lane
  int j = e & 15;              // half index within lane
  int K = ((L >> 4) << 4) + j; // 0..31 within the 32-deep K tile
  int c = L & 15;              // column within the 16-wide N tile
  float v;
  if (f < 8) {
    int k = f >> 2, n = f & 3;
    v = w0[(k * 32 + K) * WID + n * 16 + c];
  } else if (f < 16) {
    int g = f - 8, k = g >> 2, n = g & 3;
    v = w1[(k * 32 + K) * WID + n * 16 + c];
  } else {
    int k = f - 16;
    v = (c < 3) ? w2[(k * 32 + K) * 3 + c] : 0.0f;
  }
  frags[idx] = (_Float16)v;
}

// ---------------------------------------------------------------------------
// Helpers
// ---------------------------------------------------------------------------
__device__ __forceinline__ v8f wmma_f16(v16h a, v16h b, v8f c) {
  return __builtin_amdgcn_wmma_f32_16x16x32_f16(
      /*neg_a=*/false, a, /*neg_b=*/false, b,
      /*c_mod=*/(short)0, c, /*reuse_a=*/false, /*reuse_b=*/false);
}

// relu in a single v_med3: med3(x, 0, +inf) = clamp(x, 0, inf)
__device__ __forceinline__ float relu1(float x) {
  return __builtin_amdgcn_fmed3f(x, 0.0f, __builtin_inff());
}

// B fragment: 16 contiguous halves per lane (loaded once from global, kept in VGPRs)
__device__ __forceinline__ v16h load_B(const _Float16* __restrict__ frags,
                                       int frag, int lane) {
  const _Float16* p = frags + frag * FRAG_HALVES + lane * 16;
  union { v16h h; uint4 u[2]; } r;
  r.u[0] = *(const uint4*)(p);
  r.u[1] = *(const uint4*)(p + 8);
  return r.h;
}

// A fragment from row-major LDS activation buffer (contiguous k order, same as B packing):
// lane<16 -> row=lane,    K = {0..15} + 32*ktile
// lane>=16-> row=lane-16, K = {16..31} + 32*ktile
__device__ __forceinline__ v16h load_A(const _Float16* hb, int lane, int ktile) {
  int row  = lane & 15;
  int koff = ktile * 32 + ((lane >> 4) << 4);
  const _Float16* base = hb + row * HSTR + koff;
  union { v16h h; uint4 u[2]; } r;
  r.u[0] = *(const uint4*)(base);       // K .. K+7
  r.u[1] = *(const uint4*)(base + 8);   // K+8 .. K+15
  return r.h;
}

// ---------------------------------------------------------------------------
// Main fused kernel: one 16-splat tile per wave per iteration.
// Weights + biases live in registers for the whole kernel; LDS is a per-wave
// activation staging buffer only (in-order DS => no barriers needed).
// WMMA accumulators start from a single shared zero vector; biases are added
// in the epilogue to keep VGPR count < 256.
// ---------------------------------------------------------------------------
__global__ __launch_bounds__(256) void
view_mlp_kernel(const float* __restrict__ features,
                const int*   __restrict__ camera_ids,
                const float* __restrict__ means,
                const float* __restrict__ colors,
                const float* __restrict__ c2w,
                const int*   __restrict__ shdeg,
                const float* __restrict__ embed_table,
                const float* __restrict__ b0,
                const float* __restrict__ b1,
                const float* __restrict__ b2,
                const _Float16* __restrict__ frags,
                float* __restrict__ out) {
  __shared__ __align__(16) _Float16 hbuf[8][16 * HSTR];   // 18.4 KB, per-wave slices

  const int tid  = threadIdx.x;
  const int wave = tid >> 5;
  const int lane = tid & 31;
  const int cl   = lane & 15;          // column within a 16-wide N tile
  const int rb   = (lane >> 4) << 3;   // C/D row base for this lane half
  _Float16* hb = &hbuf[wave][0];

  // ---- resident weights: 18 fragments = 144 VGPRs ----
  v16h Bf[NFRAG];
#pragma unroll
  for (int f = 0; f < NFRAG; ++f) Bf[f] = load_B(frags, f, lane);

  // ---- resident per-lane bias scalars (added in epilogue, not in C) ----
  float bias0[4], bias1[4];
#pragma unroll
  for (int n = 0; n < 4; ++n) {
    bias0[n] = b0[n * 16 + cl];
    bias1[n] = b1[n * 16 + cl];
  }
  const float bias2 = (cl < 3) ? b2[cl] : 0.0f;

  const int deg = shdeg[0];
  const int nb  = (deg + 1) * (deg + 1);

  // single zero accumulator seed shared by all WMMAs (8 VGPRs)
  v8f zacc = {};

  for (int tile = blockIdx.x * 8 + wave; tile < NTILES; tile += gridDim.x * 8) {
    const int cam = tile / TILES_PER_CAM;
    const int s0  = (tile - cam * TILES_PER_CAM) * 16;
    const float cpx = c2w[cam * 16 + 3];
    const float cpy = c2w[cam * 16 + 7];
    const float cpz = c2w[cam * 16 + 11];
    const int cid = camera_ids[cam];

    // ---- prefetch next tile's features ----
    {
      int nt = tile + gridDim.x * 8;
      if (nt < NTILES) {
        int ncam = nt / TILES_PER_CAM;
        int ns0  = (nt - ncam * TILES_PER_CAM) * 16;
        __builtin_prefetch(features + (size_t)(ns0 + (lane >> 1)) * FDIM +
                               (lane & 1) * 16, 0, 1);
      }
    }

    // ---- build input h = [embed(16) | features(32) | sh(16)] in f16 LDS ----
    {   // features: lane L -> row L/2, cols (L&1)*16..+15
      int row = lane >> 1, cb = (lane & 1) * 16;
      const float* fp = features + (size_t)(s0 + row) * FDIM + cb;
      _Float16* dst = hb + row * HSTR + EDIM + cb;
#pragma unroll
      for (int j = 0; j < 16; j += 4) {
        float4 v = *(const float4*)(fp + j);
        dst[j + 0] = (_Float16)v.x; dst[j + 1] = (_Float16)v.y;
        dst[j + 2] = (_Float16)v.z; dst[j + 3] = (_Float16)v.w;
      }
    }
    if (lane < 16) {   // SH bases for row = lane
      int row = lane, splat = s0 + row;
      float dx = means[splat * 3 + 0] - cpx;
      float dy = means[splat * 3 + 1] - cpy;
      float dz = means[splat * 3 + 2] - cpz;
      float inv = rsqrtf(fmaxf(dx * dx + dy * dy + dz * dz, 1e-24f));
      float x = dx * inv, y = dy * inv, z = dz * inv;
      float z2  = z * z;
      float fC1 = x * x - y * y, fS1 = 2.0f * x * y;
      float fC2 = x * fC1 - y * fS1, fS2 = x * fS1 + y * fC1;
      const float fTmp0B = -1.092548430592079f;
      float fTmp0C = -2.285228997322329f * z2 + 0.4570457994644658f;
      float fTmp1B = 1.445305721320277f * z;
      float s[16];
      s[0]  = 0.2820947917738781f;
      s[1]  = -0.48860251190292f * y;
      s[2]  = 0.48860251190292f * z;
      s[3]  = -0.48860251190292f * x;
      s[4]  = 0.5462742152960395f * fS1;
      s[5]  = fTmp0B * y * z;
      s[6]  = 0.9461746957575601f * z2 - 0.3153915652525201f;
      s[7]  = fTmp0B * x * z;
      s[8]  = 0.5462742152960395f * fC1;
      s[9]  = -0.5900435899266435f * fS2;
      s[10] = fTmp1B * fS1;
      s[11] = fTmp0C * y;
      s[12] = z * (1.865881662950577f * z2 - 1.119528997770346f);
      s[13] = fTmp0C * x;
      s[14] = fTmp1B * fC1;
      s[15] = -0.5900435899266435f * fC2;
      _Float16* dst = hb + row * HSTR + EDIM + FDIM;
#pragma unroll
      for (int i = 0; i < 16; ++i) dst[i] = (_Float16)((i < nb) ? s[i] : 0.0f);
    } else {           // embed broadcast for row = lane-16
      int row = lane - 16;
      const float* ep = embed_table + (size_t)cid * EDIM;
      _Float16* dst = hb + row * HSTR;
#pragma unroll
      for (int j = 0; j < 16; ++j) dst[j] = (_Float16)ep[j];
    }

    // ---- layer 0: 16x64 @ 64x64, bias+relu in epilogue ----
    v16h a0 = load_A(hb, lane, 0);
    v16h a1 = load_A(hb, lane, 1);
    {
      v8f acc[4];
#pragma unroll
      for (int n = 0; n < 4; ++n) {
        v8f c = wmma_f16(a0, Bf[0 * 4 + n], zacc);
        acc[n] = wmma_f16(a1, Bf[1 * 4 + n], c);
      }
#pragma unroll
      for (int n = 0; n < 4; ++n)
#pragma unroll
        for (int v = 0; v < 8; ++v)
          hb[(rb + v) * HSTR + n * 16 + cl] =
              (_Float16)relu1(acc[n][v] + bias0[n]);
    }

    // ---- layer 1: 16x64 @ 64x64, bias+relu in epilogue ----
    a0 = load_A(hb, lane, 0);
    a1 = load_A(hb, lane, 1);
    {
      v8f acc[4];
#pragma unroll
      for (int n = 0; n < 4; ++n) {
        v8f c = wmma_f16(a0, Bf[8 + 0 * 4 + n], zacc);
        acc[n] = wmma_f16(a1, Bf[8 + 1 * 4 + n], c);
      }
#pragma unroll
      for (int n = 0; n < 4; ++n)
#pragma unroll
        for (int v = 0; v < 8; ++v)
          hb[(rb + v) * HSTR + n * 16 + cl] =
              (_Float16)relu1(acc[n][v] + bias1[n]);
    }

    // ---- layer 2: 16x64 @ 64x3 (N padded to 16), sigmoid ----
    a0 = load_A(hb, lane, 0);
    a1 = load_A(hb, lane, 1);
    {
      v8f c = wmma_f16(a0, Bf[16], zacc);
      c = wmma_f16(a1, Bf[17], c);

      // sigmoid into per-wave LDS float scratch (hbuf is free now),
      // then fully-coalesced colors-add + store of 48 contiguous floats.
      float* fsc = (float*)hb;
      if (cl < 3) {
#pragma unroll
        for (int v = 0; v < 8; ++v)
          fsc[(rb + v) * 3 + cl] = 1.0f / (1.0f + __expf(-(c[v] + bias2)));
      }
#pragma unroll
      for (int i = 0; i < 2; ++i) {
        int idx = lane + i * 32;
        if (idx < 48) {
          float val = fsc[idx] + colors[(size_t)s0 * 3 + idx];
          out[((size_t)cam * NSPL + s0) * 3 + idx] = val;
        }
      }
    }
  }
}

// ---------------------------------------------------------------------------
extern "C" void kernel_launch(void* const* d_in, const int* in_sizes, int n_in,
                              void* d_out, int out_size, void* d_ws, size_t ws_size,
                              hipStream_t stream) {
  const float* features = (const float*)d_in[0];
  const int*   camids   = (const int*)  d_in[1];
  const float* means    = (const float*)d_in[2];
  const float* colors   = (const float*)d_in[3];
  const float* c2w      = (const float*)d_in[4];
  const int*   shdeg    = (const int*)  d_in[5];
  const float* embed    = (const float*)d_in[6];
  const float* w0       = (const float*)d_in[7];
  const float* b0       = (const float*)d_in[8];
  const float* w1       = (const float*)d_in[9];
  const float* b1       = (const float*)d_in[10];
  const float* w2       = (const float*)d_in[11];
  const float* b2       = (const float*)d_in[12];
  float* out = (float*)d_out;
  _Float16* frags = (_Float16*)d_ws;   // 18*512*2 = 18432 bytes of scratch

  prep_weights_kernel<<<(NFRAG * FRAG_HALVES + 255) / 256, 256, 0, stream>>>(
      w0, w1, w2, frags);
  view_mlp_kernel<<<2048, 256, 0, stream>>>(
      features, camids, means, colors, c2w, shdeg, embed,
      b0, b1, b2, frags, out);
}
